// Spatial_Basic_Block_79869211836747
// MI455X (gfx1250) — matrix operand order repaired and checked by reading, not compiled, not run on hardware
//
#include <hip/hip_runtime.h>
#include <hip/hip_bf16.h>

// ST-GCN spatial block, fused:  out = relu( sum_k (W_k @ x) @ A_k + bias_term + x )
// bf16 WMMA (f32 accumulate) on gfx1250, wave32. W/X/Y staged in LDS as bf16.

typedef __bf16 bf16;
typedef __attribute__((ext_vector_type(16))) bf16  v16bf;
typedef __attribute__((ext_vector_type(8)))  bf16  bf16x8;
typedef __attribute__((ext_vector_type(4)))  bf16  bf16x4;
typedef __attribute__((ext_vector_type(8)))  float v8f;
typedef __attribute__((ext_vector_type(4)))  float f32x4;

#define C_IN   64
#define C_OUT  64
#define KK     3
#define TDIM   256
#define VDIM   25
#define TT     8                 // timesteps per block
#define PP     (TT*VDIM)         // 200 real positions per block
#define PPAD   208               // padded to 13 WMMA N-tiles
#define NT_P   (PPAD/16)         // 13
#define XSTR   72                // LDS row stride (bf16), padded vs 64 for banks
#define NWAVE  8

__device__ __forceinline__ v16bf make_frag(bf16x8 lo, bf16x8 hi) {
  union { v16bf v; bf16x8 h[2]; } u;
  u.h[0] = lo; u.h[1] = hi;
  return u.v;
}

// A-fragment (16x32 bf16) from an LDS row-major matrix (row stride XSTR):
// lane m=lane&15, half=lane>>4; values are 16 contiguous elements at
// K = kc0 + half*16 + 0..15  -> two ds_load_b128.
__device__ __forceinline__ v16bf load_amat_frag(const bf16* __restrict__ M,
                                                int r0, int kc0, int lane) {
  int m = lane & 15, half = lane >> 4;
  const bf16* base = M + (r0 + m) * XSTR + kc0 + half * 16;
  return make_frag(*(const bf16x8*)base, *(const bf16x8*)(base + 8));
}

// B-fragment (32x16 bf16) of X from LDS [p][c]: lane = column p0+(lane&15),
// K = kc0 + (lane>>4)*16 + 0..15 contiguous -> two ds_load_b128.
__device__ __forceinline__ v16bf load_x_frag(const bf16* __restrict__ Xs,
                                             int p0, int kc0, int lane) {
  int col = p0 + (lane & 15);
  int kb  = kc0 + (lane >> 4) * 16;
  const bf16* base = Xs + col * XSTR + kb;
  return make_frag(*(const bf16x8*)base, *(const bf16x8*)(base + 8));
}

// A-fragment (16x32 bf16) of Y from LDS [c][t][32]: tile rows = (c,t),
// K = v (zero padded 25..31) -> two ds_load_b128.
__device__ __forceinline__ v16bf load_y_frag(const bf16* __restrict__ Ys,
                                             int mt, int lane) {
  int ri = lane & 15, half = lane >> 4;
  int c = 2 * mt + (ri >> 3), t = ri & 7;
  const bf16* row = Ys + (c * TT + t) * 32 + half * 16;
  return make_frag(*(const bf16x8*)row, *(const bf16x8*)(row + 8));
}

// B-fragment (32x16 bf16) of A_k (v x w), zero-padded to 32x32; gathered from global.
__device__ __forceinline__ v16bf load_a_frag(const float* __restrict__ Ag,
                                             int k, int nt, int lane) {
  int w  = (lane & 15) + nt * 16;
  int vb = (lane >> 4) * 16;
  union { v16bf v; bf16 e[16]; } u;
#pragma unroll
  for (int j = 0; j < 16; ++j) {
    int vv = vb + j;
    float av = (vv < VDIM && w < VDIM) ? Ag[(k * VDIM + vv) * VDIM + w] : 0.0f;
    u.e[j] = (bf16)av;
  }
  return u.v;
}

__global__ __launch_bounds__(256)
void stgcn_fused(const float* __restrict__ xg, const float* __restrict__ Ag,
                 const float* __restrict__ Wg, const float* __restrict__ bg,
                 float* __restrict__ outg) {
  __shared__ __align__(16) bf16  Xs[PPAD * XSTR];        // X^T tile, bf16 [p][c]
  __shared__ __align__(16) bf16  Ys[C_OUT * TT * 32];    // Y_k, bf16 [c][t][v32]
  __shared__ __align__(16) bf16  Ws[KK * C_OUT * XSTR];  // W, bf16 [o][c]
  __shared__ float cs[KK * 32];                          // colsum(A_k)[w]
  __shared__ float bs[KK * C_OUT];                       // bias

  const int tid  = threadIdx.x;
  const int lane = tid & 31;
  const int wave = tid >> 5;
  const int tt   = blockIdx.x;           // 0..31  (t tile)
  const int nb   = blockIdx.y;           // 0..63  (batch)
  const int t0   = tt * TT;
  const size_t xbase = (size_t)nb * C_IN * TDIM * VDIM;
  const float* xn = xg + xbase + (size_t)t0 * VDIM;

  // ---- prologue: stage W (bf16), X^T (bf16), bias, A column sums into LDS
  for (int idx = tid; idx < (KK * C_OUT * C_IN) / 4; idx += 256) {
    int o  = idx >> 4;                   // 16 float4 per row of 64
    int c4 = (idx & 15) * 4;
    f32x4 wv = *(const f32x4*)(Wg + o * C_IN + c4);
    union { bf16x4 v; bf16 e[4]; } uu;
    uu.e[0] = (bf16)wv.x; uu.e[1] = (bf16)wv.y;
    uu.e[2] = (bf16)wv.z; uu.e[3] = (bf16)wv.w;
    *(bf16x4*)(&Ws[o * XSTR + c4]) = uu.v;
  }
  for (int idx = tid; idx < C_IN * (PPAD / 4); idx += 256) {
    int c = idx / (PPAD / 4), q = idx - c * (PPAD / 4);
    int p = q * 4;
    f32x4 xv = {0.f, 0.f, 0.f, 0.f};
    if (p < PP) xv = *(const f32x4*)(xn + (size_t)c * (TDIM * VDIM) + p);
    Xs[(p + 0) * XSTR + c] = (bf16)xv.x;
    Xs[(p + 1) * XSTR + c] = (bf16)xv.y;
    Xs[(p + 2) * XSTR + c] = (bf16)xv.z;
    Xs[(p + 3) * XSTR + c] = (bf16)xv.w;
  }
  // Zero Ys once (v = 25..31 padding must stay 0 for every k).
  for (int idx = tid; idx < (C_OUT * TT * 32) / 2; idx += 256)
    ((unsigned int*)Ys)[idx] = 0u;
  if (tid < KK * 32) {                   // colsum(A_k)[w]
    int k = tid >> 5, w = tid & 31;
    float s = 0.f;
    if (w < VDIM)
      for (int v = 0; v < VDIM; ++v) s += Ag[(k * VDIM + v) * VDIM + w];
    cs[tid] = s;
  }
  if (tid < KK * C_OUT) bs[tid] = bg[tid];
  __syncthreads();

  const v8f zacc = {0.f, 0.f, 0.f, 0.f, 0.f, 0.f, 0.f, 0.f};
  v8f acc[4][2];
#pragma unroll
  for (int i = 0; i < 4; ++i) { acc[i][0] = zacc; acc[i][1] = zacc; }

  for (int k = 0; k < KK; ++k) {
    // ---- stage 1: Y_k = W_k @ X  ->  Ys[c][t][v]
    // W fragments for this k held in registers (4 o-tiles x 2 K-chunks).
    v16bf wa[4][2];
#pragma unroll
    for (int ot = 0; ot < 4; ++ot) {
      wa[ot][0] = load_amat_frag(Ws, k * C_OUT + ot * 16, 0,  lane);
      wa[ot][1] = load_amat_frag(Ws, k * C_OUT + ot * 16, 32, lane);
    }
    for (int pt = wave; pt < NT_P; pt += NWAVE) {
      int p0 = pt * 16;
      v16bf b0 = load_x_frag(Xs, p0, 0,  lane);
      v16bf b1 = load_x_frag(Xs, p0, 32, lane);
      int col  = p0 + (lane & 15);
      int t    = col / VDIM, v = col - t * VDIM;
      int half = lane >> 4;
      bool live = (t < TT);              // padded positions (p>=200) dropped
#pragma unroll
      for (int ot = 0; ot < 4; ++ot) {
        v8f d = zacc;
        d = __builtin_amdgcn_wmma_f32_16x16x32_bf16(false, wa[ot][0], false, b0, (short)0, d, false, false);
        d = __builtin_amdgcn_wmma_f32_16x16x32_bf16(false, wa[ot][1], false, b1, (short)0, d, false, false);
        if (live) {
          // D layout: M = r + 8*(lane>>4); c = ot*16 + M; rows are 512B apart.
          bf16* dst = &Ys[((ot * 16 + 8 * half) * TT + t) * 32 + v];
#pragma unroll
          for (int r = 0; r < 8; ++r) dst[r * TT * 32] = (bf16)d[r];
        }
      }
    }
    __syncthreads();

    // ---- stage 2: acc[(c,t), w] += Y_k[(c,t), v] @ A_k[v, w]
    v16bf bA0 = load_a_frag(Ag, k, 0, lane);
    v16bf bA1 = load_a_frag(Ag, k, 1, lane);
#pragma unroll
    for (int i = 0; i < 4; ++i) {
      int mt = wave * 4 + i;             // M-tile = 2 channels x 8 timesteps
      v16bf ya = load_y_frag(Ys, mt, lane);
      acc[i][0] = __builtin_amdgcn_wmma_f32_16x16x32_bf16(false, ya, false, bA0, (short)0, acc[i][0], false, false);
      acc[i][1] = __builtin_amdgcn_wmma_f32_16x16x32_bf16(false, ya, false, bA1, (short)0, acc[i][1], false, false);
    }
    __syncthreads();                     // Ys reused by next k's stage 1
  }

  // ---- epilogue: out = relu(acc + bias_term(c,w) + x), fp32
  const int half = lane >> 4;
  const int wl   = lane & 15;
#pragma unroll
  for (int i = 0; i < 4; ++i) {
    int mt = wave * 4 + i;
    int c  = 2 * mt + half;              // row = c*8 + t; M = r + 8*half
#pragma unroll
    for (int nt = 0; nt < 2; ++nt) {
      int w = wl + nt * 16;
      if (w < VDIM) {
        float bterm = bs[c]             * cs[w]
                    + bs[C_OUT + c]     * cs[32 + w]
                    + bs[2 * C_OUT + c] * cs[64 + w];
#pragma unroll
        for (int r = 0; r < 8; ++r) {
          size_t gi = xbase + ((size_t)c * TDIM + (t0 + r)) * VDIM + w;
          float val = acc[i][nt][r] + bterm + xg[gi];
          outg[gi] = val > 0.0f ? val : 0.0f;
        }
      }
    }
  }
}

extern "C" void kernel_launch(void* const* d_in, const int* in_sizes, int n_in,
                              void* d_out, int out_size, void* d_ws, size_t ws_size,
                              hipStream_t stream) {
  const float* x = (const float*)d_in[0];   // [64,64,256,25]
  const float* A = (const float*)d_in[1];   // [3,25,25]
  const float* W = (const float*)d_in[2];   // [192,64]
  const float* b = (const float*)d_in[3];   // [192]
  float* out = (float*)d_out;               // out (26214400) ++ A (1875)

  dim3 grid(TDIM / TT, 64);                 // 32 t-tiles x 64 batches
  stgcn_fused<<<grid, 256, 0, stream>>>(x, A, W, b, out);

  // Second tuple element: pass A through.
  hipMemcpyAsync(out + (size_t)64 * C_OUT * TDIM * VDIM, A,
                 (size_t)KK * VDIM * VDIM * sizeof(float),
                 hipMemcpyDeviceToDevice, stream);
}
